// MolGAN_Discriminator_32298154066175
// MI455X (gfx1250) — matrix operand order repaired and checked by reading, not compiled, not run on hardware
//
#include <hip/hip_runtime.h>

// MolGAN discriminator for MI455X (gfx1250, wave32, WMMA).
//
// Pipeline (all exact algebra, f16 WMMA with f32 accumulate):
//   kP : Wc = W1[:, :128] @ W0  ([64,16]), dA = W1a@b0 ; WcB = Wc + W1b, dB = dA + b1
//   k1 : y0 = node @ Wc.T + dA ; c1 = node @ WcB.T + dB        [B,N,64] each
//   k2 : x1 = adj @ y0 + c1      (WMMA, y0^T in LDS as f16)    [B,N,64]
//   k3 : h1 = adj @ x1 + x1 ; ann=[h1,node] ; i=sig(ann Ws^T+bs), j=tanh(ann Wt^T+bt)
//        partial[b][slab][:] = sum_rows i*j    (all WMMA + LDS, fused epilogue)
//   k4 : g=tanh(sum partials); g=tanh(g Wl^T+bl); out = g Wo^T + bo   [B,1]

#define Bb   128
#define Nn   1024
#define KP   1032   // padded LDS stride (halves) for the 64x1024 f16 matrix
#define KP3  104    // padded LDS stride (halves) for 80->96 K-dim matrices

typedef __attribute__((ext_vector_type(16))) _Float16 v16h;
typedef __attribute__((ext_vector_type(8)))  float    v8f;
typedef __attribute__((ext_vector_type(4)))  float    f4;

__device__ __forceinline__ v8f wmma_f16(v16h a, v16h b, v8f c) {
  // (neg_a, A, neg_b, B, c_mod, C, reuse_a, reuse_b)
  return __builtin_amdgcn_wmma_f32_16x16x32_f16(false, a, false, b, (short)0, c,
                                                false, false);
}

// A fragment from global f32 row: runs of 8 at +0 and +16 floats (per ISA A layout):
// a[0..7] = A[M][kb+aoff .. +7], a[8..15] = A[M][kb+16+aoff .. +7]
__device__ __forceinline__ v16h load_a_gl(const float* __restrict__ p) {
  f4 r0 = *(const f4*)(p);
  f4 r1 = *(const f4*)(p + 4);
  f4 r2 = *(const f4*)(p + 16);
  f4 r3 = *(const f4*)(p + 20);
  v16h a;
#pragma unroll
  for (int i = 0; i < 4; ++i) {
    a[i]      = (_Float16)r0[i];
    a[4 + i]  = (_Float16)r1[i];
    a[8 + i]  = (_Float16)r2[i];
    a[12 + i] = (_Float16)r3[i];
  }
  return a;
}

// 16 contiguous halves (B fragment: K ascending for one column)
__device__ __forceinline__ v16h load_h16(const _Float16* p) {
  union { f4 f[2]; v16h h; } u;
  u.f[0] = *(const f4*)(p);
  u.f[1] = *(const f4*)(p + 8);
  return u.h;
}

// A fragment from LDS f16 row: 8 halves at +0 and 8 halves at +16
__device__ __forceinline__ v16h load_a_lds(const _Float16* p) {
  union { f4 f[2]; v16h h; } u;
  u.f[0] = *(const f4*)(p);
  u.f[1] = *(const f4*)(p + 16);
  return u.h;
}

__device__ __forceinline__ float fast_tanh(float x) {
  x = fminf(fmaxf(x, -15.f), 15.f);
  float e = __expf(2.f * x);
  return (e - 1.f) / (e + 1.f);
}
__device__ __forceinline__ float fast_sigmoid(float x) {
  return 1.f / (1.f + __expf(-x));
}

// ---------------- kP: fold the first two linear layers into [64,16] weights --
__global__ __launch_bounds__(256) void kP(const float* __restrict__ W0,
                                          const float* __restrict__ b0,
                                          const float* __restrict__ W1,
                                          const float* __restrict__ b1,
                                          float* __restrict__ wcA, float* __restrict__ dA,
                                          float* __restrict__ wcB, float* __restrict__ dB) {
  int t = threadIdx.x;
  for (int o = t; o < 64 * 16; o += 256) {
    int j = o >> 4, f = o & 15;
    float s = 0.f;
    for (int k = 0; k < 128; ++k) s += W1[j * 144 + k] * W0[k * 16 + f];
    wcA[o] = s;
    wcB[o] = s + W1[j * 144 + 128 + f];
  }
  if (t < 64) {
    float d = 0.f;
    for (int k = 0; k < 128; ++k) d += W1[t * 144 + k] * b0[k];
    dA[t] = d;
    dB[t] = d + b1[t];
  }
}

// ---------------- k1: y0 / c1 from node with folded weights ------------------
__global__ __launch_bounds__(256) void k1(const float* __restrict__ node,
                                          const float* __restrict__ wcA,
                                          const float* __restrict__ dA,
                                          const float* __restrict__ wcB,
                                          const float* __restrict__ dB,
                                          float* __restrict__ y0,
                                          float* __restrict__ cc1) {
  __shared__ float wA[1024], wB[1024], dAl[64], dBl[64], nr[4][16];
  int tid = threadIdx.x;
  for (int i = tid; i < 1024; i += 256) { wA[i] = wcA[i]; wB[i] = wcB[i]; }
  if (tid < 64) { dAl[tid] = dA[tid]; dBl[tid] = dB[tid]; }
  size_t nb = (size_t)blockIdx.x * 4;  // 4 graph-nodes per block
  if (tid < 64) nr[tid >> 4][tid & 15] = node[nb * 16 + tid];
  __syncthreads();
  int nl = tid >> 6, j = tid & 63;
  float sA = dAl[j], sB = dBl[j];
#pragma unroll
  for (int f = 0; f < 16; ++f) {
    float nv = nr[nl][f];
    sA += wA[j * 16 + f] * nv;
    sB += wB[j * 16 + f] * nv;
  }
  size_t o = (nb + nl) * 64 + j;
  y0[o] = sA;
  cc1[o] = sB;
}

// ---------------- k2: x1 = adj @ y0 + c1 (WMMA) ------------------------------
// grid = (N/128, B), 256 threads (8 waves); wave w owns a 16-row strip.
__global__ __launch_bounds__(256) void k2(const float* __restrict__ adj,
                                          const float* __restrict__ y0,
                                          const float* __restrict__ cc1,
                                          float* __restrict__ x1) {
  __shared__ _Float16 XT[64 * KP];  // y0[b]^T as f16, K-major per column
  const int b = blockIdx.y, slab = blockIdx.x;
  const int tid = threadIdx.x;
  for (int idx = tid; idx < Nn * 64; idx += 256) {
    int n = idx >> 6, c = idx & 63;
    XT[c * KP + n] = (_Float16)y0[((size_t)b * Nn + n) * 64 + c];
  }
  __syncthreads();
  const int l = tid & 31, w = tid >> 5;
  const int m = l & 15;
  const int hi = (l >> 4) & 1;
  const int aoff = hi * 8;    // A-fragment K sub-offset for upper half-wave
  const int boff = hi * 8;    // B-fragment K sub-offset for upper half-wave
  const int growA = slab * 128 + w * 16 + m;
  const float* arow = adj + ((size_t)b * Nn + growA) * Nn + aoff;
  v8f acc[4] = {};
  for (int kb = 0; kb < Nn; kb += 32) {
    v16h a = load_a_gl(arow + kb);
    __builtin_prefetch(arow + kb + 256, 0, 1);  // -> global_prefetch_b8
#pragma unroll
    for (int g = 0; g < 4; ++g) {
      v16h bm = load_a_lds(&XT[(16 * g + m) * KP + kb + boff]);
      acc[g] = wmma_f16(a, bm, acc[g]);
    }
  }
#pragma unroll
  for (int g = 0; g < 4; ++g) {
#pragma unroll
    for (int r = 0; r < 8; ++r) {
      int grow = slab * 128 + w * 16 + r + hi * 8;
      int col = 16 * g + m;
      size_t o = ((size_t)b * Nn + grow) * 64 + col;
      x1[o] = acc[g][r] + cc1[o];
    }
  }
}

// ---------------- k3: h1 = adj@x1 + x1 ; fused gated aggregation -------------
__global__ __launch_bounds__(256) void k3(const float* __restrict__ adj,
                                          const float* __restrict__ x1,
                                          const float* __restrict__ node,
                                          const float* __restrict__ Ws,
                                          const float* __restrict__ bs,
                                          const float* __restrict__ Wt,
                                          const float* __restrict__ bt,
                                          float* __restrict__ gpart) {
  __shared__ _Float16 XT[64 * KP];      // x1[b]^T f16
  __shared__ _Float16 WsK[128 * KP3];   // Ws K-major, K padded 80->104 (zeros)
  __shared__ _Float16 WtK[128 * KP3];
  __shared__ _Float16 annL[128 * KP3];  // [h1 | node | 0], row-major, K padded
  __shared__ float bsL[128], btL[128];
  __shared__ float wp[8][128];
  const int b = blockIdx.y, slab = blockIdx.x;
  const int tid = threadIdx.x;

  for (int idx = tid; idx < Nn * 64; idx += 256) {
    int n = idx >> 6, c = idx & 63;
    XT[c * KP + n] = (_Float16)x1[((size_t)b * Nn + n) * 64 + c];
  }
  for (int idx = tid; idx < 128 * KP3; idx += 256) {
    int col = idx / KP3, k = idx - col * KP3;
    _Float16 zs = (_Float16)((k < 80) ? Ws[col * 80 + k] : 0.f);
    _Float16 zt = (_Float16)((k < 80) ? Wt[col * 80 + k] : 0.f);
    WsK[idx] = zs;
    WtK[idx] = zt;
  }
  for (int idx = tid; idx < 128 * 40; idx += 256) {  // ann cols 64..103
    int row = idx / 40, k = 64 + (idx - row * 40);
    float v = (k < 80) ? node[((size_t)b * Nn + slab * 128 + row) * 16 + (k - 64)] : 0.f;
    annL[row * KP3 + k] = (_Float16)v;
  }
  if (tid < 128) { bsL[tid] = bs[tid]; btL[tid] = bt[tid]; }
  __syncthreads();

  const int l = tid & 31, w = tid >> 5;
  const int m = l & 15;
  const int hi = (l >> 4) & 1;
  const int aoff = hi * 8;
  const int boff = hi * 8;

  // ---- stage 1: h1 strip = adj @ x1 (+ x1), written into annL cols 0..63 ----
  {
    const int growA = slab * 128 + w * 16 + m;
    const float* arow = adj + ((size_t)b * Nn + growA) * Nn + aoff;
    v8f acc[4] = {};
    for (int kb = 0; kb < Nn; kb += 32) {
      v16h a = load_a_gl(arow + kb);
      __builtin_prefetch(arow + kb + 256, 0, 1);
#pragma unroll
      for (int g = 0; g < 4; ++g) {
        v16h bm = load_a_lds(&XT[(16 * g + m) * KP + kb + boff]);
        acc[g] = wmma_f16(a, bm, acc[g]);
      }
    }
#pragma unroll
    for (int g = 0; g < 4; ++g) {
#pragma unroll
      for (int r = 0; r < 8; ++r) {
        int rl = w * 16 + r + hi * 8;             // local row in slab
        int col = 16 * g + m;
        float h = acc[g][r] + (float)XT[col * KP + slab * 128 + rl];
        annL[rl * KP3 + col] = (_Float16)h;
      }
    }
  }
  __syncthreads();

  // ---- stage 2: i/j = ann @ {Ws,Wt}^T ; s = sum_rows sigmoid(i)*tanh(j) -----
  {
    const int arow2 = w * 16 + m;
    v8f ia[8] = {};
    v8f ja[8] = {};
#pragma unroll
    for (int kb = 0; kb < 96; kb += 32) {
      v16h a = load_a_lds(&annL[arow2 * KP3 + kb + aoff]);
#pragma unroll
      for (int g = 0; g < 8; ++g) {
        v16h bS = load_a_lds(&WsK[(16 * g + m) * KP3 + kb + boff]);
        v16h bT = load_a_lds(&WtK[(16 * g + m) * KP3 + kb + boff]);
        ia[g] = wmma_f16(a, bS, ia[g]);
        ja[g] = wmma_f16(a, bT, ja[g]);
      }
    }
#pragma unroll
    for (int g = 0; g < 8; ++g) {
      int col = 16 * g + m;
      float s = 0.f;
#pragma unroll
      for (int r = 0; r < 8; ++r) {
        float iv = fast_sigmoid(ia[g][r] + bsL[col]);
        float jv = fast_tanh(ja[g][r] + btL[col]);
        s += iv * jv;
      }
      s += __shfl_xor(s, 16, 32);  // combine row halves (same col in lane l, l+16)
      if (l < 16) wp[w][16 * g + l] = s;
    }
  }
  __syncthreads();
  if (tid < 128) {
    float t = 0.f;
#pragma unroll
    for (int w2 = 0; w2 < 8; ++w2) t += wp[w2][tid];  // fixed order: deterministic
    gpart[((size_t)b * 8 + slab) * 128 + tid] = t;
  }
}

// ---------------- k4: readout --------------------------------------------------
__global__ __launch_bounds__(128) void k4(const float* __restrict__ gpart,
                                          const float* __restrict__ Wl,
                                          const float* __restrict__ bl,
                                          const float* __restrict__ Wo,
                                          const float* __restrict__ bo,
                                          float* __restrict__ out) {
  __shared__ float gl[128];
  __shared__ float t2[64];
  int b = blockIdx.x, t = threadIdx.x;
  float s = 0.f;
#pragma unroll
  for (int sl = 0; sl < 8; ++sl) s += gpart[((size_t)b * 8 + sl) * 128 + t];
  gl[t] = fast_tanh(s);
  __syncthreads();
  if (t < 64) {
    float a = bl[t];
    for (int c = 0; c < 128; ++c) a += Wl[t * 128 + c] * gl[c];
    t2[t] = fast_tanh(a);
  }
  __syncthreads();
  if (t == 0) {
    float a = bo[0];
    for (int j = 0; j < 64; ++j) a += Wo[j] * t2[j];
    out[b] = a;
  }
}

extern "C" void kernel_launch(void* const* d_in, const int* in_sizes, int n_in,
                              void* d_out, int out_size, void* d_ws, size_t ws_size,
                              hipStream_t stream) {
  const float* node = (const float*)d_in[0];
  const float* adj  = (const float*)d_in[1];
  const float* W0   = (const float*)d_in[2];
  const float* b0   = (const float*)d_in[3];
  const float* W1   = (const float*)d_in[4];
  const float* b1   = (const float*)d_in[5];
  const float* Ws   = (const float*)d_in[6];
  const float* bs   = (const float*)d_in[7];
  const float* Wt   = (const float*)d_in[8];
  const float* bt   = (const float*)d_in[9];
  const float* Wl   = (const float*)d_in[10];
  const float* bl   = (const float*)d_in[11];
  const float* Wo   = (const float*)d_in[12];
  const float* bo   = (const float*)d_in[13];

  float* ws = (float*)d_ws;
  const size_t NODES = (size_t)Bb * Nn;           // 131072
  float* wcA   = ws;                               // 1024
  float* dA    = ws + 1024;                        // 64
  float* wcB   = ws + 1088;                        // 1024
  float* dB    = ws + 2112;                        // 64
  float* y0    = ws + 4096;                        // NODES*64
  float* cc1   = y0 + NODES * 64;                  // NODES*64
  float* x1    = cc1 + NODES * 64;                 // NODES*64
  float* gpart = x1 + NODES * 64;                  // B*8*128
  // total ~101 MB of workspace

  kP<<<1, 256, 0, stream>>>(W0, b0, W1, b1, wcA, dA, wcB, dB);
  k1<<<(unsigned)(NODES / 4), 256, 0, stream>>>(node, wcA, dA, wcB, dB, y0, cc1);
  k2<<<dim3(Nn / 128, Bb), 256, 0, stream>>>(adj, y0, cc1, x1);
  k3<<<dim3(Nn / 128, Bb), 256, 0, stream>>>(adj, x1, node, Ws, bs, Wt, bt, gpart);
  k4<<<Bb, 128, 0, stream>>>(gpart, Wl, bl, Wo, bo, (float*)d_out);
}
